// StaticSparseAttention_38920993636560
// MI455X (gfx1250) — matrix operand order repaired and checked by reading, not compiled, not run on hardware
//
#include <hip/hip_runtime.h>
#include <hip/hip_bf16.h>
#include <math.h>

typedef __bf16 v8bf  __attribute__((ext_vector_type(8)));
typedef __bf16 v16bf __attribute__((ext_vector_type(16)));
typedef float  v8f   __attribute__((ext_vector_type(8)));
typedef float  v4f   __attribute__((ext_vector_type(4)));

#define DEVINL __device__ __forceinline__

// Flip to 0 if the async-LDS asm is rejected by this toolchain.
#ifndef USE_ASYNC_LDS
#define USE_ASYNC_LDS 1
#endif

static constexpr int kB    = 2;
static constexpr int kH    = 8;
static constexpr int kHD   = 64;
static constexpr int kE    = 512;
static constexpr int kS    = 2049;      // NCLS + P
static constexpr int kP    = 2048;
static constexpr int kQKVN = 3 * kE;    // 1536
static constexpr float kScale = 0.125f; // 64^-0.5

DEVINL v16bf cat8(v8bf lo, v8bf hi) {
  return __builtin_shufflevector(lo, hi, 0,1,2,3,4,5,6,7,8,9,10,11,12,13,14,15);
}
DEVINL v8f wmma_bf16_f32(v16bf a, v16bf b, v8f c) {
  return __builtin_amdgcn_wmma_f32_16x16x32_bf16(false, a, false, b, (short)0, c,
                                                 false, false);
}

// 16-byte global -> LDS async copy (ASYNCcnt-tracked on CDNA5).
// LDS operand = low 32 bits of flat pointer (aperture truncation, ISA 10.2).
DEVINL void async_copy_b128(__bf16* lds_dst, const __bf16* gsrc) {
#if USE_ASYNC_LDS
  asm volatile("global_load_async_to_lds_b128 %0, %1, off"
               :: "v"((unsigned)(size_t)lds_dst), "v"(gsrc)
               : "memory");
#else
  *(v8bf*)lds_dst = *(const v8bf*)gsrc;
#endif
}
DEVINL void async_wait_all() {
#if USE_ASYNC_LDS
  asm volatile("s_wait_asynccnt 0x0" ::: "memory");
#endif
}

// ---------------------------------------------------------------------------
// One-shot f32 -> bf16 conversion (x, qkv_w, out_w).
// ---------------------------------------------------------------------------
__global__ __launch_bounds__(256)
void cvt_f32_to_bf16(const float* __restrict__ src, __bf16* __restrict__ dst,
                     int n) {
  const int i = (blockIdx.x * 256 + threadIdx.x) * 4;
  if (i + 3 < n) {
    const v4f v = *(const v4f*)(src + i);
#pragma unroll
    for (int j = 0; j < 4; ++j) dst[i + j] = (__bf16)v[j];
  } else {
    for (int j = 0; j < 4 && i + j < n; ++j) dst[i + j] = (__bf16)src[i + j];
  }
}

// ---------------------------------------------------------------------------
// C = A(MxK,bf16) * W(NxK,bf16)^T + bias(f32), OutT in {float,__bf16}.
// 256 threads / 8 waves; 128x128 tile; wave computes 32x64 via 8 WMMA accums.
// Tiles staged to LDS with async global->LDS b128 copies.
// Requires K % 32 == 0, N % 128 == 0; M guarded.
// ---------------------------------------------------------------------------
static constexpr int GT_M = 128, GT_N = 128, GT_K = 32, GLD = 40;

template <typename OutT>
__global__ __launch_bounds__(256)
void gemm_bf16(const __bf16* __restrict__ A, const __bf16* __restrict__ W,
               const float* __restrict__ bias, OutT* __restrict__ C,
               int M, int N, int K) {
  __shared__ __bf16 As[GT_M * GLD];
  __shared__ __bf16 Ws[GT_N * GLD];

  const int tid  = threadIdx.x;
  const int lane = tid & 31;
  const int wave = tid >> 5;
  const int hf   = lane >> 4;
  const int ln   = lane & 15;
  const int wm   = wave >> 1;
  const int wn   = wave & 1;
  const int rowB = blockIdx.x * GT_M;
  const int colB = blockIdx.y * GT_N;

  const v8f vzero = {0.f,0.f,0.f,0.f,0.f,0.f,0.f,0.f};
  v8f acc[2][4];
#pragma unroll
  for (int s = 0; s < 2; ++s)
#pragma unroll
    for (int t = 0; t < 4; ++t) acc[s][t] = vzero;

  const int frow = tid >> 1;        // stage row 0..127
  const int fcol = (tid & 1) * 16;  // 0 or 16

  for (int k0 = 0; k0 < K; k0 += GT_K) {
    // --- stage A tile ---
    {
      const int gr = rowB + frow;
      __bf16* dst = &As[frow * GLD + fcol];
      if (gr < M) {
        const __bf16* src = A + (size_t)gr * K + k0 + fcol;
        if (k0 + GT_K < K) __builtin_prefetch(src + GT_K, 0, 1);
        async_copy_b128(dst, src);
        async_copy_b128(dst + 8, src + 8);
      } else {
#pragma unroll
        for (int i = 0; i < 16; ++i) dst[i] = (__bf16)0.f;
      }
    }
    // --- stage W tile (N multiple of 128 -> no guard) ---
    {
      const __bf16* src = W + (size_t)(colB + frow) * K + k0 + fcol;
      if (k0 + GT_K < K) __builtin_prefetch(src + GT_K, 0, 1);
      __bf16* dst = &Ws[frow * GLD + fcol];
      async_copy_b128(dst, src);
      async_copy_b128(dst + 8, src + 8);
    }
    async_wait_all();
    __syncthreads();

    // A frag: lane = row; lanes0-15 K=0..15, lanes16-31 K=16..31
    v16bf af[2];
#pragma unroll
    for (int s = 0; s < 2; ++s) {
      const __bf16* p = &As[(wm * 32 + s * 16 + ln) * GLD + hf * 16];
      af[s] = cat8(*(const v8bf*)p, *(const v8bf*)(p + 8));
    }
    // B frag: lane = col; lanes0-15 K=0..15, lanes16-31 K=16..31
#pragma unroll
    for (int t = 0; t < 4; ++t) {
      const __bf16* p = &Ws[(wn * 64 + t * 16 + ln) * GLD + hf * 16];
      v16bf bf = cat8(*(const v8bf*)p, *(const v8bf*)(p + 8));
#pragma unroll
      for (int s = 0; s < 2; ++s)
        acc[s][t] = wmma_bf16_f32(af[s], bf, acc[s][t]);
    }
    __syncthreads();
  }

#pragma unroll
  for (int t = 0; t < 4; ++t) {
    const int col = colB + wn * 64 + t * 16 + ln;
    const float bv = bias[col];
#pragma unroll
    for (int s = 0; s < 2; ++s) {
#pragma unroll
      for (int r = 0; r < 8; ++r) {
        const int row = rowB + wm * 32 + s * 16 + r + 8 * hf;
        if (row < M) C[(size_t)row * N + col] = (OutT)(acc[s][t][r] + bv);
      }
    }
  }
}

// ---------------------------------------------------------------------------
// CLS query attends to all S keys (bf16 qkv, f32 math, bf16 ctx out). Tiny.
// ---------------------------------------------------------------------------
__global__ __launch_bounds__(256)
void cls_attn(const __bf16* __restrict__ qkv, __bf16* __restrict__ ctx) {
  __shared__ float qv[kHD];
  __shared__ float sc[kS];
  __shared__ float red[256];
  __shared__ float part[4][kHD];

  const int tid = threadIdx.x;
  const int h = blockIdx.x % kH;
  const int b = blockIdx.x / kH;

  const __bf16* qc = qkv + (size_t)b * kS * kQKVN + h * kHD;
  if (tid < kHD) qv[tid] = (float)qc[tid];
  __syncthreads();

  float lmax = -3.0e38f;
  for (int s = tid; s < kS; s += 256) {
    const __bf16* kr = qkv + ((size_t)b * kS + s) * kQKVN + kE + h * kHD;
    float d = 0.f;
#pragma unroll 8
    for (int i = 0; i < kHD; ++i) d += qv[i] * (float)kr[i];
    d *= kScale;
    sc[s] = d;
    lmax = fmaxf(lmax, d);
  }
  red[tid] = lmax;
  __syncthreads();
  for (int off = 128; off >= 1; off >>= 1) {
    if (tid < off) red[tid] = fmaxf(red[tid], red[tid + off]);
    __syncthreads();
  }
  const float mx = red[0];
  __syncthreads();

  float lsum = 0.f;
  for (int s = tid; s < kS; s += 256) {
    const float e = __expf(sc[s] - mx);
    sc[s] = e;
    lsum += e;
  }
  red[tid] = lsum;
  __syncthreads();
  for (int off = 128; off >= 1; off >>= 1) {
    if (tid < off) red[tid] += red[tid + off];
    __syncthreads();
  }
  const float esum = red[0];

  const int d = tid & (kHD - 1);
  const int g = tid >> 6;
  float a = 0.f;
  for (int s = g; s < kS; s += 4) {
    const __bf16* vr = qkv + ((size_t)b * kS + s) * kQKVN + 2 * kE + h * kHD;
    a += sc[s] * (float)vr[d];
  }
  part[g][d] = a;
  __syncthreads();
  if (tid < kHD) {
    const float o = part[0][tid] + part[1][tid] + part[2][tid] + part[3][tid];
    ctx[(size_t)b * kS * kE + h * kHD + tid] = (__bf16)(o / esum);
  }
}

// ---------------------------------------------------------------------------
// Dilated-window attention via dense WMMA tiles (bf16 qkv in, bf16 ctx out).
// 64 threads = 2 waves; wave owns 16 consecutive queries.
// Key rows p0-64..p0+95 (160) + CLS (160) + zero (161) staged in LDS;
// K row-major via async b128 copies, V transposed through VGPRs.
// ---------------------------------------------------------------------------
static constexpr int AW   = 2;
static constexpr int WQ   = 32;
static constexpr int NC   = 160;
static constexpr int KR   = 162;
static constexpr int KSTR = 72;   // 144B rows (16B-aligned)
static constexpr int VSTR = 176;  // cols 162..175 zeroed

__global__ __launch_bounds__(64)
void win_attn(const __bf16* __restrict__ qkv, __bf16* __restrict__ ctx) {
  __shared__ __bf16 Ks[KR * KSTR];      // [key row][d]
  __shared__ __bf16 Vt[kHD * VSTR];     // [d][key row]
  __shared__ __bf16 Ab[AW * 16 * NC];   // per-wave unnormalized probs

  const int tid = threadIdx.x;
  const int nb  = kP / WQ;
  const int pb  = blockIdx.x % nb;
  const int h   = (blockIdx.x / nb) % kH;
  const int b   = blockIdx.x / (nb * kH);
  const int p0  = pb * WQ;

  const __bf16* qbase = qkv + ((size_t)b * kS + 1) * kQKVN + h * kHD;
  const __bf16* kbase = qbase + kE;
  const __bf16* vbase = qbase + 2 * kE;
  const __bf16* kcls  = qkv + (size_t)b * kS * kQKVN + kE + h * kHD;
  const __bf16* vcls  = kcls + kE;

  const v8bf bzero = {(__bf16)0.f,(__bf16)0.f,(__bf16)0.f,(__bf16)0.f,
                      (__bf16)0.f,(__bf16)0.f,(__bf16)0.f,(__bf16)0.f};

  // ---- stage K rows (async) and transposed V ----
  for (int e = tid; e < KR * 8; e += 64) {
    const int r   = e >> 3;
    const int sg  = (e & 7) * 8;
    const int pos = p0 - 64 + r;
    __bf16* kd = &Ks[r * KSTR + sg];
    v8bf vv;
    if (r == 160) {
      async_copy_b128(kd, kcls + sg);
      vv = *(const v8bf*)(vcls + sg);
    } else if (r == 161 || pos < 0 || pos >= kP) {
      *(v8bf*)kd = bzero;
      vv = bzero;
    } else {
      async_copy_b128(kd, kbase + (size_t)pos * kQKVN + sg);
      vv = *(const v8bf*)(vbase + (size_t)pos * kQKVN + sg);
    }
#pragma unroll
    for (int i = 0; i < 8; ++i) Vt[(sg + i) * VSTR + r] = vv[i];
  }
  for (int e = tid; e < kHD * (VSTR - KR); e += 64) {
    const int d = e / (VSTR - KR);
    const int c = KR + e % (VSTR - KR);
    Vt[d * VSTR + c] = (__bf16)0.f;
  }
  async_wait_all();
  __syncthreads();

  const int lane = tid & 31;
  const int w    = tid >> 5;
  const int hf   = lane >> 4;
  const int ln   = lane & 15;

  // ---- Q fragments: direct 16B bf16 loads from global ----
  v16bf aq[2];
  {
    const __bf16* qr = qbase + (size_t)(p0 + w * 16 + ln) * kQKVN;
#pragma unroll
    for (int kk = 0; kk < 2; ++kk) {
      const int d0 = kk * 32 + hf * 16;
      aq[kk] = cat8(*(const v8bf*)(qr + d0), *(const v8bf*)(qr + d0 + 8));
    }
  }

  // ---- scores: 10 column tiles x 2 k-steps ----
  const v8f vzero = {0.f,0.f,0.f,0.f,0.f,0.f,0.f,0.f};
  v8f sc[10];
#pragma unroll
  for (int t = 0; t < 10; ++t) sc[t] = vzero;
#pragma unroll
  for (int t = 0; t < 10; ++t) {
    const int kr = (t < 9) ? (w * 16 + t * 16 + ln) : ((ln == 0) ? 160 : 161);
#pragma unroll
    for (int kk = 0; kk < 2; ++kk) {
      const __bf16* p = &Ks[kr * KSTR + kk * 32 + hf * 16];
      v16bf bk = cat8(*(const v8bf*)p, *(const v8bf*)(p + 8));
      sc[t] = wmma_bf16_f32(aq[kk], bk, sc[t]);
    }
  }

  // ---- mask + scale: rel = j - m in [0,128] even; pos in [0,P) ----
  const float NEG = -__builtin_inff();
#pragma unroll
  for (int t = 0; t < 10; ++t) {
#pragma unroll
    for (int r = 0; r < 8; ++r) {
      const int m = r + 8 * hf;
      bool valid;
      if (t < 9) {
        const int j   = t * 16 + ln;
        const int rel = j - m;
        const int pos = p0 + w * 16 - 64 + j;
        valid = (rel >= 0) & (rel <= 128) & ((rel & 1) == 0) &
                (pos >= 0) & (pos < kP);
      } else {
        valid = (ln == 0);  // CLS column j == 144
      }
      sc[t][r] = valid ? sc[t][r] * kScale : NEG;
    }
  }

  // ---- row softmax via xor butterflies over the 16-lane group ----
  float rsum[8];
#pragma unroll
  for (int r = 0; r < 8; ++r) {
    float mxv = sc[0][r];
#pragma unroll
    for (int t = 1; t < 10; ++t) mxv = fmaxf(mxv, sc[t][r]);
#pragma unroll
    for (int d = 1; d < 16; d <<= 1) mxv = fmaxf(mxv, __shfl_xor(mxv, d, 32));
    float s = 0.f;
#pragma unroll
    for (int t = 0; t < 10; ++t) {
      const float e = __expf(sc[t][r] - mxv);
      sc[t][r] = e;
      s += e;
    }
#pragma unroll
    for (int d = 1; d < 16; d <<= 1) s += __shfl_xor(s, d, 32);
    rsum[r] = s;
  }

  // ---- restripe probs C-layout -> A-layout through LDS ----
  __bf16* abw = &Ab[w * 16 * NC];
#pragma unroll
  for (int t = 0; t < 10; ++t) {
#pragma unroll
    for (int r = 0; r < 8; ++r) {
      const int m = r + 8 * hf;
      abw[m * NC + t * 16 + ln] = (__bf16)sc[t][r];
    }
  }

  // ---- O = A(16x160) x V(160x64) ----
  v8f oacc[4];
#pragma unroll
  for (int dt = 0; dt < 4; ++dt) oacc[dt] = vzero;
#pragma unroll
  for (int kk = 0; kk < 5; ++kk) {
    const __bf16* pa = &abw[ln * NC + kk * 32 + hf * 16];
    const v16bf afr = cat8(*(const v8bf*)pa, *(const v8bf*)(pa + 8));
    const int jst = kk * 32 + hf * 16;
    const int rst = (jst < 144) ? (w * 16 + jst) : (160 + jst - 144);
#pragma unroll
    for (int dt = 0; dt < 4; ++dt) {
      const __bf16* pv = &Vt[(dt * 16 + ln) * VSTR + rst];
      const v16bf bfr = cat8(*(const v8bf*)pv, *(const v8bf*)(pv + 8));
      oacc[dt] = wmma_bf16_f32(afr, bfr, oacc[dt]);
    }
  }

  // ---- normalize, write bf16 ctx[b, 1+p, h*64+d] ----
#pragma unroll
  for (int dt = 0; dt < 4; ++dt) {
#pragma unroll
    for (int r = 0; r < 8; ++r) {
      const int m = r + 8 * hf;
      const size_t row = (size_t)b * kS + 1 + p0 + w * 16 + m;
      ctx[row * kE + h * kHD + dt * 16 + ln] = (__bf16)(oacc[dt][r] / rsum[r]);
    }
  }
}

// ---------------------------------------------------------------------------
extern "C" void kernel_launch(void* const* d_in, const int* in_sizes, int n_in,
                              void* d_out, int out_size, void* d_ws, size_t ws_size,
                              hipStream_t stream) {
  (void)in_sizes; (void)n_in; (void)out_size; (void)ws_size;
  const float* x     = (const float*)d_in[0];
  const float* qkv_w = (const float*)d_in[1];
  const float* qkv_b = (const float*)d_in[2];
  const float* out_w = (const float*)d_in[3];
  const float* out_b = (const float*)d_in[4];
  float* out = (float*)d_out;

  // bf16 workspace: qkv | ctx | x | qkv_w | out_w  (all 16B-multiple sizes)
  __bf16* qkv_bf  = (__bf16*)d_ws;
  __bf16* ctx_bf  = qkv_bf  + (size_t)kB * kS * kQKVN;
  __bf16* x_bf    = ctx_bf  + (size_t)kB * kS * kE;
  __bf16* qkvw_bf = x_bf    + (size_t)kB * kS * kE;
  __bf16* outw_bf = qkvw_bf + (size_t)kQKVN * kE;

  const int M   = kB * kS;            // 4098
  const int nx  = kB * kS * kE;       // x elements
  const int nqw = kQKVN * kE;
  const int now = kE * kE;

  cvt_f32_to_bf16<<<dim3((nx / 4 + 255) / 256), dim3(256), 0, stream>>>(x, x_bf, nx);
  cvt_f32_to_bf16<<<dim3((nqw / 4 + 255) / 256), dim3(256), 0, stream>>>(qkv_w, qkvw_bf, nqw);
  cvt_f32_to_bf16<<<dim3((now / 4 + 255) / 256), dim3(256), 0, stream>>>(out_w, outw_bf, now);

  dim3 blk(256);
  dim3 g1((M + GT_M - 1) / GT_M, kQKVN / GT_N);
  gemm_bf16<__bf16><<<g1, blk, 0, stream>>>(x_bf, qkvw_bf, qkv_b, qkv_bf,
                                            M, kQKVN, kE);

  cls_attn<<<dim3(kB * kH), dim3(256), 0, stream>>>(qkv_bf, ctx_bf);
  win_attn<<<dim3(kB * kH * (kP / WQ)), dim3(64), 0, stream>>>(qkv_bf, ctx_bf);

  dim3 g2((M + GT_M - 1) / GT_M, kE / GT_N);
  gemm_bf16<float><<<g2, blk, 0, stream>>>(ctx_bf, outw_bf, out_b, out,
                                           M, kE, kE);
}
